// Coconut_66168266162345
// MI455X (gfx1250) — compile-verified, matrix operand order checked by reading, not run on hardware
//
#include <hip/hip_runtime.h>
#include <hip/hip_bf16.h>
#include <cstdint>

// ---------------- problem constants ----------------
constexpr int CB = 4;       // batch
constexpr int CS = 512;     // max seq
constexpr int CD = 1024;    // model dim
constexpr int CH = 16;      // heads
constexpr int CL = 4;       // layers
constexpr int CV = 32000;   // vocab
constexpr int CDH = 64;     // head dim

typedef __attribute__((ext_vector_type(16))) __bf16 v16bf;
typedef __attribute__((ext_vector_type(8)))  __bf16 v8bf;
typedef __attribute__((ext_vector_type(8)))  float  v8f;
typedef __attribute__((ext_vector_type(4)))  unsigned int v4u;
typedef __attribute__((ext_vector_type(8)))  int v8i_;
typedef __attribute__((ext_vector_type(4)))  int v4i_;

#if __has_builtin(__builtin_amdgcn_tensor_load_to_lds)
#define USE_TDM 1
#define AS_PAD 0
#else
#define USE_TDM 0
#define AS_PAD 8
#endif

// Build a 16x32 bf16 A/B fragment register block from two 16-byte LDS chunks.
// Lane layout (ISA 7.12.2, 16-bit A 16x32): lanes 0-15 hold K=0..7 (elems 0-7)
// and K=16..23 (elems 8-15); lanes 16-31 hold K=8..15 and K=24..31.
__device__ inline v16bf make_frag(const __bf16* lo, const __bf16* hi) {
  const v8bf a = *(const v8bf*)lo;
  const v8bf b = *(const v8bf*)hi;
  v16bf r;
#pragma unroll
  for (int i = 0; i < 8; ++i) { r[i] = a[i]; r[8 + i] = b[i]; }
  return r;
}

// ---------------- generic bf16 GEMM:  C(f32)[M,N] = A(bf16)[M,K] * B(bf16)[K,N]
#define GT_M 128
#define GT_N 64
#define GT_K 32

__global__ __launch_bounds__(256) void gemm_bf16_kernel(
    const __bf16* __restrict__ A, const __bf16* __restrict__ B,
    float* __restrict__ C, int M, int N, int K, int lda, int ldb, int ldc) {
  __shared__ __bf16 As[GT_M][GT_K + AS_PAD];  // row-major A tile
  __shared__ __bf16 Bs[GT_N][GT_K + 8];       // stored transposed: [n][k]
  const int bm = blockIdx.y * GT_M;
  const int bn = blockIdx.x * GT_N;
  const int tid = threadIdx.x;
  const int lane = tid & 31, wave = tid >> 5;
  const int wm = (wave & 3) * 32;   // 4 waves tile M
  const int wn = (wave >> 2) * 32;  // 2 waves tile N
  const int mhalf = lane & 15, khalf = lane >> 4;
  v8f c[2][2] = {};
#if USE_TDM
  const unsigned as_lds = (unsigned)(uintptr_t)&As[0][0];  // LDS address = low 32 bits of the shared pointer
#endif

  for (int k0 = 0; k0 < K; k0 += GT_K) {
#if USE_TDM
    // --- stage A via Tensor Data Mover: 128x32 bf16 2D tile, zero-filled
    // beyond tensor_dim (handles the M tail with no VALU guards).
    if (wave == 0) {
      const unsigned long long ga =
          (unsigned long long)(uintptr_t)(A + (size_t)bm * lda + k0);
      v4u g0;
      g0[0] = 1u;                                   // count=1, user mode
      g0[1] = as_lds;                               // lds_addr
      g0[2] = (unsigned)(ga & 0xffffffffu);         // global_addr[31:0]
      g0[3] = (unsigned)((ga >> 32) & 0x1ffffffu) | (2u << 30);  // addr[56:32] | type=2
      const unsigned td0 = (unsigned)(K - k0);      // valid elems in dim0 from tile start
      const unsigned td1 = (unsigned)(M - bm);      // valid rows -> OOB rows read 0
      v8i_ g1;
      g1[0] = 0x10000;                              // data_size = 2 bytes
      g1[1] = (int)((td0 & 0xffffu) << 16);         // tensor_dim0[15:0]
      g1[2] = (int)(((td0 >> 16) & 0xffffu) | ((td1 & 0xffffu) << 16));
      g1[3] = (int)(((td1 >> 16) & 0xffffu) | ((unsigned)GT_K << 16));  // tile_dim0=32
      g1[4] = GT_M;                                 // tile_dim1=128 (tile_dim2=0)
      g1[5] = lda;                                  // tensor_dim0_stride[31:0]
      g1[6] = 0;                                    // stride hi + dim1_stride lo
      g1[7] = 0;
      v4i_ gz = {};
      v8i_ gz8 = {};
      __builtin_amdgcn_tensor_load_to_lds(g0, g1, gz, gz, gz8, 0);
    }
#else
    // --- stage A: 128 x 32, each thread 16 contiguous elems
    {
      const int r = tid >> 1;
      const int cb = (tid & 1) * 16;
      const int gr = bm + r;
      if (gr < M) {
        const v8bf* src = (const v8bf*)(A + (size_t)gr * lda + k0 + cb);
        *(v8bf*)&As[r][cb]     = src[0];
        *(v8bf*)&As[r][cb + 8] = src[1];
      } else {
        v8bf z = {};
        *(v8bf*)&As[r][cb] = z; *(v8bf*)&As[r][cb + 8] = z;
      }
    }
#endif
    // --- stage B: 32 x 64, store transposed [n][k]
    {
      const int kk = tid >> 3;
      const int nb = (tid & 7) * 8;
      const v8bf src = *(const v8bf*)(B + (size_t)(k0 + kk) * ldb + bn + nb);
#pragma unroll
      for (int i = 0; i < 8; ++i) Bs[nb + i][kk] = src[i];
      if (k0 + GT_K < K)
        __builtin_prefetch(B + (size_t)(k0 + GT_K + kk) * ldb + bn + nb, 0, 1);
    }
#if USE_TDM
    if (wave == 0) __builtin_amdgcn_s_wait_tensorcnt(0);
#endif
    __syncthreads();
#pragma unroll
    for (int mt = 0; mt < 2; ++mt) {
      const v16bf a = make_frag(&As[wm + mt * 16 + mhalf][khalf * 8],
                                &As[wm + mt * 16 + mhalf][16 + khalf * 8]);
#pragma unroll
      for (int nt = 0; nt < 2; ++nt) {
        const v16bf b = make_frag(&Bs[wn + nt * 16 + mhalf][khalf * 8],
                                  &Bs[wn + nt * 16 + mhalf][16 + khalf * 8]);
        c[mt][nt] = __builtin_amdgcn_wmma_f32_16x16x32_bf16(
            false, a, false, b, (short)0, c[mt][nt], false, false);
      }
    }
    __syncthreads();
  }
  // --- store C: VGPR i -> row (khalf*8 + i), col = mhalf  (ISA C/D layout)
#pragma unroll
  for (int mt = 0; mt < 2; ++mt)
#pragma unroll
    for (int nt = 0; nt < 2; ++nt) {
      const int col = bn + wn + nt * 16 + mhalf;
#pragma unroll
      for (int i = 0; i < 8; ++i) {
        const int row = bm + wm + mt * 16 + khalf * 8 + i;
        if (row < M) C[(size_t)row * ldc + col] = c[mt][nt][i];
      }
    }
}

// ---------------- attention: scores tile = Q @ K^T (causal, scaled) ----------
__global__ __launch_bounds__(32) void attn_scores_kernel(
    const float* __restrict__ qkv, float* __restrict__ scores, int T) {
  const int kt = blockIdx.x, qt = blockIdx.y, bh = blockIdx.z;
  const int b = bh / CH, h = bh % CH;
  const int q0 = qt * 16, k0 = kt * 16;
  float* sbase = scores + (size_t)bh * T * T;
  const int lane = threadIdx.x;

  if (k0 > q0 + 15) {  // fully masked tile
    for (int idx = lane; idx < 256; idx += 32) {
      const int q = q0 + (idx >> 4), k = k0 + (idx & 15);
      if (q < T && k < T) sbase[(size_t)q * T + k] = -1e9f;
    }
    return;
  }
  __shared__ __bf16 Qs[16][72];
  __shared__ __bf16 Ks[16][72];
  {
    const int r = lane >> 1;
    const int db = (lane & 1) * 32;
    const int q = q0 + r, k = k0 + r;
    const float* sq = qkv + ((size_t)(b * T + q) * 3 * CD) + h * CDH + db;
    const float* sk = qkv + ((size_t)(b * T + k) * 3 * CD) + CD + h * CDH + db;
#pragma unroll 8
    for (int i = 0; i < 32; ++i) {
      Qs[r][db + i] = (q < T) ? (__bf16)sq[i] : (__bf16)0.f;
      Ks[r][db + i] = (k < T) ? (__bf16)sk[i] : (__bf16)0.f;
    }
  }
  __syncthreads();
  const int mhalf = lane & 15, khalf = lane >> 4;
  v8f c = {};
#pragma unroll
  for (int ks = 0; ks < 2; ++ks) {
    const v16bf a = make_frag(&Qs[mhalf][ks * 32 + khalf * 8],
                              &Qs[mhalf][ks * 32 + 16 + khalf * 8]);
    const v16bf b = make_frag(&Ks[mhalf][ks * 32 + khalf * 8],
                              &Ks[mhalf][ks * 32 + 16 + khalf * 8]);
    c = __builtin_amdgcn_wmma_f32_16x16x32_bf16(false, a, false, b, (short)0, c,
                                                false, false);
  }
  const float scale = 0.125f;  // 1/sqrt(64)
#pragma unroll
  for (int i = 0; i < 8; ++i) {
    const int q = q0 + khalf * 8 + i;
    const int k = k0 + mhalf;
    if (q < T && k < T)
      sbase[(size_t)q * T + k] = (k <= q) ? c[i] * scale : -1e9f;
  }
}

// ---------------- row softmax over scores ----------------
__global__ __launch_bounds__(128) void softmax_rows_kernel(float* __restrict__ sc, int T) {
  float* p = sc + (size_t)blockIdx.x * T;
  __shared__ float redm[4], reds[4];
  float mx = -1e30f;
  for (int i = threadIdx.x; i < T; i += 128) mx = fmaxf(mx, p[i]);
  for (int o = 16; o > 0; o >>= 1) mx = fmaxf(mx, __shfl_xor(mx, o, 32));
  if ((threadIdx.x & 31) == 0) redm[threadIdx.x >> 5] = mx;
  __syncthreads();
  mx = fmaxf(fmaxf(redm[0], redm[1]), fmaxf(redm[2], redm[3]));
  float s = 0.f;
  for (int i = threadIdx.x; i < T; i += 128) {
    const float e = expf(p[i] - mx);
    p[i] = e; s += e;
  }
  for (int o = 16; o > 0; o >>= 1) s += __shfl_xor(s, o, 32);
  if ((threadIdx.x & 31) == 0) reds[threadIdx.x >> 5] = s;
  __syncthreads();
  s = reds[0] + reds[1] + reds[2] + reds[3];
  const float inv = 1.f / s;
  for (int i = threadIdx.x; i < T; i += 128) p[i] *= inv;
}

// ---------------- attention: O = P @ V, writes bf16 (B,T,D) ----------------
__global__ __launch_bounds__(32) void attn_v_kernel(
    const float* __restrict__ qkv, const float* __restrict__ scores,
    __bf16* __restrict__ o_bf, int T) {
  const int qt = blockIdx.x, bh = blockIdx.y;
  const int b = bh / CH, h = bh % CH;
  const int q0 = qt * 16;
  __shared__ __bf16 Ps[16][40];
  __shared__ __bf16 Vs[64][40];   // transposed: [d][k]
  const int lane = threadIdx.x;
  const int mhalf = lane & 15, khalf = lane >> 4;
  v8f c[4] = {};
  const int kmax = min(T, q0 + 16);
  for (int k0 = 0; k0 < kmax; k0 += 32) {
    {
      const int r = lane >> 1, cb = (lane & 1) * 16;
      const int q = q0 + r;
      const float* src = scores + ((size_t)bh * T + q) * T + k0 + cb;
#pragma unroll 4
      for (int i = 0; i < 16; ++i) {
        const int kk = k0 + cb + i;
        Ps[r][cb + i] = (q < T && kk < T) ? (__bf16)src[i] : (__bf16)0.f;
      }
    }
    {
      const int kk = k0 + lane;
      const float* src = qkv + ((size_t)(b * T + kk) * 3 * CD) + 2 * CD + h * CDH;
      const bool ok = kk < T;
#pragma unroll 8
      for (int d = 0; d < 64; ++d)
        Vs[d][lane] = ok ? (__bf16)src[d] : (__bf16)0.f;
    }
    __syncthreads();
    const v16bf a = make_frag(&Ps[mhalf][khalf * 8], &Ps[mhalf][16 + khalf * 8]);
#pragma unroll
    for (int nt = 0; nt < 4; ++nt) {
      const v16bf bb = make_frag(&Vs[nt * 16 + mhalf][khalf * 8],
                                 &Vs[nt * 16 + mhalf][16 + khalf * 8]);
      c[nt] = __builtin_amdgcn_wmma_f32_16x16x32_bf16(false, a, false, bb,
                                                      (short)0, c[nt], false, false);
    }
    __syncthreads();
  }
#pragma unroll
  for (int nt = 0; nt < 4; ++nt)
#pragma unroll
    for (int i = 0; i < 8; ++i) {
      const int q = q0 + khalf * 8 + i;
      if (q < T)
        o_bf[(size_t)(b * T + q) * CD + h * CDH + nt * 16 + mhalf] = (__bf16)c[nt][i];
    }
}

// ---------------- layernorm: one block per row ----------------
__global__ __launch_bounds__(256) void layernorm_kernel(
    const float* __restrict__ x, const float* __restrict__ g,
    const float* __restrict__ bta, __bf16* __restrict__ out_bf,
    float* __restrict__ out_f) {
  const int row = blockIdx.x;
  const float* xr = x + (size_t)row * CD;
  __shared__ float r0[8], r1[8], mv[2];
  float s = 0.f, ss = 0.f;
  for (int i = threadIdx.x; i < CD; i += 256) {
    const float v = xr[i];
    s += v; ss += v * v;
  }
  for (int o = 16; o > 0; o >>= 1) { s += __shfl_xor(s, o, 32); ss += __shfl_xor(ss, o, 32); }
  if ((threadIdx.x & 31) == 0) { r0[threadIdx.x >> 5] = s; r1[threadIdx.x >> 5] = ss; }
  __syncthreads();
  if (threadIdx.x == 0) {
    float a = 0.f, b2 = 0.f;
    for (int i = 0; i < 8; ++i) { a += r0[i]; b2 += r1[i]; }
    const float m = a / CD;
    mv[0] = m;
    mv[1] = rsqrtf(b2 / CD - m * m + 1e-5f);
  }
  __syncthreads();
  const float m = mv[0], inv = mv[1];
  for (int i = threadIdx.x; i < CD; i += 256) {
    const float v = (xr[i] - m) * inv * g[i] + bta[i];
    out_bf[(size_t)row * CD + i] = (__bf16)v;
    if (out_f) out_f[(size_t)row * CD + i] = v;
  }
}

// ---------------- small elementwise kernels ----------------
__global__ void convert_bf16_kernel(const float* __restrict__ in,
                                    __bf16* __restrict__ out, int n) {
  const int i = blockIdx.x * 256 + threadIdx.x;
  if (i < n) out[i] = (__bf16)in[i];
}
__global__ void add_inplace_kernel(float* __restrict__ x,
                                   const float* __restrict__ y, int n) {
  const int i = blockIdx.x * 256 + threadIdx.x;
  if (i < n) x[i] += y[i];
}
__global__ void gelu_bf16_kernel(const float* __restrict__ in,
                                 __bf16* __restrict__ out, int n) {
  const int i = blockIdx.x * 256 + threadIdx.x;
  if (i < n) {
    const float x = in[i];
    const float t = tanhf(0.7978845608028654f * (x + 0.044715f * x * x * x));
    out[i] = (__bf16)(0.5f * x * (1.f + t));
  }
}
__global__ void embed_kernel(const int* __restrict__ ids,
                             const float* __restrict__ emb,
                             float* __restrict__ out) {
  const size_t i = (size_t)blockIdx.x * 256 + threadIdx.x;  // over CB*CS*CD
  const int tok = ids[i / CD];
  out[i] = emb[(size_t)tok * CD + (i % CD)];
}
__global__ void add_pos_kernel(float* __restrict__ x,
                               const float* __restrict__ inp_emb,
                               const float* __restrict__ pos_emb, int T) {
  const size_t i = (size_t)blockIdx.x * 256 + threadIdx.x;  // over CB*T*CD
  if (i >= (size_t)CB * T * CD) return;
  const int d = i % CD;
  const int t = (i / CD) % T;
  const int b = i / ((size_t)T * CD);
  x[i] = inp_emb[((size_t)b * CS + t) * CD + d] + pos_emb[(size_t)t * CD + d];
}
__global__ void copy_latent_kernel(float* __restrict__ inp_emb,
                                   const float* __restrict__ hid, int T, int pos) {
  const int i = blockIdx.x * 256 + threadIdx.x;  // over CB*CD
  if (i >= CB * CD) return;
  const int b = i / CD, d = i % CD;
  inp_emb[((size_t)b * CS + pos) * CD + d] = hid[((size_t)b * T + pos - 1) * CD + d];
}
__global__ void zero_kernel(float* p) { p[0] = 0.f; }

// ---------------- cross-entropy loss ----------------
__global__ __launch_bounds__(256) void ce_loss_kernel(
    const float* __restrict__ logits, const int* __restrict__ labels,
    float* __restrict__ loss) {
  const int r = blockIdx.x;               // over CB*(CS-1)
  const int b = r / (CS - 1), t = r % (CS - 1);
  const float* lp = logits + ((size_t)b * CS + t) * CV;
  const int lab = labels[b * CS + t + 1];
  __shared__ float red[8];
  float mx = -1e30f;
  for (int i = threadIdx.x; i < CV; i += 256) mx = fmaxf(mx, lp[i]);
  for (int o = 16; o > 0; o >>= 1) mx = fmaxf(mx, __shfl_xor(mx, o, 32));
  if ((threadIdx.x & 31) == 0) red[threadIdx.x >> 5] = mx;
  __syncthreads();
  mx = red[0];
  for (int i = 1; i < 8; ++i) mx = fmaxf(mx, red[i]);
  __syncthreads();
  float s = 0.f;
  for (int i = threadIdx.x; i < CV; i += 256) s += expf(lp[i] - mx);
  for (int o = 16; o > 0; o >>= 1) s += __shfl_xor(s, o, 32);
  if ((threadIdx.x & 31) == 0) red[threadIdx.x >> 5] = s;
  __syncthreads();
  if (threadIdx.x == 0) {
    float tot = 0.f;
    for (int i = 0; i < 8; ++i) tot += red[i];
    const float nll = (mx + logf(tot)) - lp[lab];
    atomicAdd(loss, nll * (1.f / (CB * (CS - 1))));
  }
}

// ---------------- host orchestration ----------------
static inline size_t align256(size_t x) { return (x + 255) & ~(size_t)255; }

extern "C" void kernel_launch(void* const* d_in, const int* in_sizes, int n_in,
                              void* d_out, int out_size, void* d_ws, size_t ws_size,
                              hipStream_t stream) {
  (void)in_sizes; (void)n_in; (void)out_size; (void)ws_size;
  // inputs (setup_inputs order)
  const int*   input_ids = (const int*)d_in[0];
  const int*   labels    = (const int*)d_in[2];
  const float* emb       = (const float*)d_in[4];
  const float* pos_emb   = (const float*)d_in[5];
  const float* Wqkv      = (const float*)d_in[6];
  const float* Wo        = (const float*)d_in[7];
  const float* W1        = (const float*)d_in[8];
  const float* W2        = (const float*)d_in[9];
  const float* ln1_g = (const float*)d_in[10];
  const float* ln1_b = (const float*)d_in[11];
  const float* ln2_g = (const float*)d_in[12];
  const float* ln2_b = (const float*)d_in[13];
  const float* lnf_g = (const float*)d_in[14];
  const float* lnf_b = (const float*)d_in[15];
  const float* lm_head = (const float*)d_in[16];

  // outputs: (loss, inputs_embeds, logits) flat
  float* loss    = (float*)d_out;
  float* inp_emb = loss + 1;                                 // CB*CS*CD
  float* logits  = inp_emb + (size_t)CB * CS * CD;           // CB*CS*CV

  // workspace bump allocator
  char* wsb = (char*)d_ws;
  size_t off = 0;
  auto alloc = [&](size_t bytes) -> void* {
    void* p = wsb + off;
    off = align256(off + bytes);
    return p;
  };
  __bf16* wqkv_bf = (__bf16*)alloc((size_t)CL * CD * 3 * CD * 2);
  __bf16* wo_bf   = (__bf16*)alloc((size_t)CL * CD * CD * 2);
  __bf16* w1_bf   = (__bf16*)alloc((size_t)CL * CD * 4 * CD * 2);
  __bf16* w2_bf   = (__bf16*)alloc((size_t)CL * 4 * CD * CD * 2);
  __bf16* lmh_bf  = (__bf16*)alloc((size_t)CD * CV * 2);
  float*  x       = (float*)alloc((size_t)CB * CS * CD * 4);
  __bf16* h_bf    = (__bf16*)alloc((size_t)CB * CS * CD * 2);
  float*  qkv     = (float*)alloc((size_t)CB * CS * 3 * CD * 4);
  float*  scores  = (float*)alloc((size_t)CB * CH * CS * CS * 4);
  __bf16* o_bf    = (__bf16*)alloc((size_t)CB * CS * CD * 2);
  float*  tmp     = (float*)alloc((size_t)CB * CS * 4 * CD * 4);
  __bf16* mlp_bf  = (__bf16*)alloc((size_t)CB * CS * 4 * CD * 2);
  float*  hid_f   = (float*)alloc((size_t)CB * CS * CD * 4);
  __bf16* hid_bf  = (__bf16*)alloc((size_t)CB * CS * CD * 2);

  auto cvt = [&](const float* src, __bf16* dst, size_t n) {
    convert_bf16_kernel<<<(unsigned)((n + 255) / 256), 256, 0, stream>>>(src, dst, (int)n);
  };
  // weights -> bf16 (166 MB total: fits the 192 MB L2, unlike fp32)
  cvt(Wqkv, wqkv_bf, (size_t)CL * CD * 3 * CD);
  cvt(Wo, wo_bf, (size_t)CL * CD * CD);
  cvt(W1, w1_bf, (size_t)CL * CD * 4 * CD);
  cvt(W2, w2_bf, (size_t)CL * 4 * CD * CD);
  cvt(lm_head, lmh_bf, (size_t)CD * CV);

  // embedding gather into the d_out inputs_embeds region (scalar accesses only)
  embed_kernel<<<(unsigned)(((size_t)CB * CS * CD) / 256), 256, 0, stream>>>(
      input_ids, emb, inp_emb);

  auto gemm = [&](const __bf16* A, const __bf16* Bw, float* C, int M, int N,
                  int K, int lda, int ldb, int ldc) {
    dim3 grid((unsigned)(N / GT_N), (unsigned)((M + GT_M - 1) / GT_M));
    gemm_bf16_kernel<<<grid, 256, 0, stream>>>(A, Bw, C, M, N, K, lda, ldb, ldc);
  };

  const int ends[7]   = {64, 65, 66, 67, 68, 69, 512};
  const int starts[7] = {0, 64, 65, 66, 67, 68, 69};

  for (int p = 0; p < 7; ++p) {
    const int T = ends[p], st = starts[p];
    const int M = CB * T;
    const int nt16 = (T + 15) / 16;

    add_pos_kernel<<<(unsigned)(((size_t)M * CD + 255) / 256), 256, 0, stream>>>(
        x, inp_emb, pos_emb, T);

    for (int l = 0; l < CL; ++l) {
      layernorm_kernel<<<M, 256, 0, stream>>>(x, ln1_g + l * CD, ln1_b + l * CD,
                                              h_bf, nullptr);
      gemm(h_bf, wqkv_bf + (size_t)l * CD * 3 * CD, qkv, M, 3 * CD, CD, CD,
           3 * CD, 3 * CD);

      attn_scores_kernel<<<dim3(nt16, nt16, CB * CH), 32, 0, stream>>>(qkv, scores, T);
      softmax_rows_kernel<<<CB * CH * T, 128, 0, stream>>>(scores, T);
      attn_v_kernel<<<dim3(nt16, CB * CH), 32, 0, stream>>>(qkv, scores, o_bf, T);

      gemm(o_bf, wo_bf + (size_t)l * CD * CD, tmp, M, CD, CD, CD, CD, CD);
      add_inplace_kernel<<<(unsigned)(((size_t)M * CD + 255) / 256), 256, 0,
                           stream>>>(x, tmp, M * CD);

      layernorm_kernel<<<M, 256, 0, stream>>>(x, ln2_g + l * CD, ln2_b + l * CD,
                                              h_bf, nullptr);
      gemm(h_bf, w1_bf + (size_t)l * CD * 4 * CD, tmp, M, 4 * CD, CD, CD,
           4 * CD, 4 * CD);
      gelu_bf16_kernel<<<(unsigned)(((size_t)M * 4 * CD + 255) / 256), 256, 0,
                         stream>>>(tmp, mlp_bf, M * 4 * CD);
      gemm(mlp_bf, w2_bf + (size_t)l * 4 * CD * CD, tmp, M, CD, 4 * CD, 4 * CD,
           CD, CD);
      add_inplace_kernel<<<(unsigned)(((size_t)M * CD + 255) / 256), 256, 0,
                           stream>>>(x, tmp, M * CD);
    }
    layernorm_kernel<<<M, 256, 0, stream>>>(x, lnf_g, lnf_b, hid_bf, hid_f);

    // lm_head on the needed slice only, per batch (rows contiguous per b)
    const int Msl = T - st;
    for (int b = 0; b < CB; ++b) {
      gemm(hid_bf + (size_t)(b * T + st) * CD, lmh_bf,
           logits + ((size_t)(b * CS + st)) * CV, Msl, CV, CD, CD, CV, CV);
    }
    if (p < 6) {
      copy_latent_kernel<<<(CB * CD + 255) / 256, 256, 0, stream>>>(
          inp_emb, hid_f, T, 64 + p);
    }
  }

  zero_kernel<<<1, 1, 0, stream>>>(loss);
  ce_loss_kernel<<<CB * (CS - 1), 256, 0, stream>>>(logits, labels, loss);
}
